// RWKV7Layer_24438363914460
// MI455X (gfx1250) — matrix operand order repaired and checked by reading, not compiled, not run on hardware
//
#include <hip/hip_runtime.h>
#include <hip/hip_bf16.h>

// ---------------------------------------------------------------------------
// RWKV7 layer for MI455X (gfx1250, wave32).
//   B=8, T=1024, C=1024, H=16, N=64, M = B*T = 8192.
// Pipeline:
//   1) convert weights fp32->bf16 (packed [Wr,Wk,Wv,Ww,Wa,Wg] + Wo)
//   2) LayerNorm + time-shift mix -> bf16 activation matrix A (M x C)
//   3) fused WMMA GEMM  A @ [W6]^T -> 6 projection buffers (epilogue applies
//      decay=exp(-exp(.)) for w, sigmoid for a/g).  Wave tile 32x64, ping-pong
//      fragment pipelining (copy-free), 8 WMMAs per k-step.
//   4) sequential RWKV state scan: 1 block (256 thr) per (b,h); state row
//      split over 4 lanes, cross-lane reductions via __shfl_xor.
//   5) GroupNorm + gate (sigmoid(g)) -> bf16 activation A2
//   6) WMMA GEMM  A2 @ Wo^T -> d_out
// ---------------------------------------------------------------------------

typedef __attribute__((ext_vector_type(16))) __bf16 v16bf;
typedef __attribute__((ext_vector_type(8)))  __bf16 v8bf;
typedef __attribute__((ext_vector_type(8)))  float  v8f;

union BF16Frag { v16bf v; v8bf h[2]; };

__device__ __forceinline__ unsigned short f2bf(float f) {
    union { float f; unsigned int u; } cv; cv.f = f;
    unsigned int u = cv.u;
    unsigned int r = u + 0x7FFFu + ((u >> 16) & 1u);   // round-to-nearest-even
    return (unsigned short)(r >> 16);
}

// ---------------------------------------------------------------------------
// fp32 -> bf16 conversion (4 elements / thread)
// ---------------------------------------------------------------------------
__global__ __launch_bounds__(256)
void f32_to_bf16_kernel(const float* __restrict__ src,
                        unsigned short* __restrict__ dst, int n4) {
    int i = blockIdx.x * 256 + threadIdx.x;
    if (i >= n4) return;
    float4 v = ((const float4*)src)[i];
    ushort4 o;
    o.x = f2bf(v.x); o.y = f2bf(v.y); o.z = f2bf(v.z); o.w = f2bf(v.w);
    ((ushort4*)dst)[i] = o;
}

// ---------------------------------------------------------------------------
// LayerNorm (biased var) + time-shift:  x_in[t] = 0.5*(xn[t] + xn[t-1])
// One block per (b,t) row; each thread handles 4 channels of rows t and t-1.
// ---------------------------------------------------------------------------
__global__ __launch_bounds__(256)
void ln_shift_kernel(const float* __restrict__ x,
                     const float* __restrict__ ln_g,
                     const float* __restrict__ ln_b,
                     unsigned short* __restrict__ Abf,
                     int T, int C) {
    const int bt  = blockIdx.x;
    const int t   = bt % T;
    const int tid = threadIdx.x;

    const float* xc = x + (size_t)bt * C;
    float4 xv = ((const float4*)xc)[tid];
    float4 pv = make_float4(0.f, 0.f, 0.f, 0.f);
    if (t > 0) pv = ((const float4*)(xc - C))[tid];

    float s0 = xv.x + xv.y + xv.z + xv.w;
    float q0 = xv.x*xv.x + xv.y*xv.y + xv.z*xv.z + xv.w*xv.w;
    float s1 = pv.x + pv.y + pv.z + pv.w;
    float q1 = pv.x*pv.x + pv.y*pv.y + pv.z*pv.z + pv.w*pv.w;

    __shared__ float4 red[256];
    red[tid] = make_float4(s0, q0, s1, q1);
    for (int off = 128; off > 0; off >>= 1) {
        __syncthreads();
        if (tid < off) {
            float4 a = red[tid], b = red[tid + off];
            a.x += b.x; a.y += b.y; a.z += b.z; a.w += b.w;
            red[tid] = a;
        }
    }
    __syncthreads();
    float4 tot = red[0];
    const float invC = 1.0f / (float)C;
    float mu0 = tot.x * invC, var0 = tot.y * invC - mu0 * mu0;
    float mu1 = tot.z * invC, var1 = tot.w * invC - mu1 * mu1;
    float rs0 = rsqrtf(var0 + 1e-5f);
    float rs1 = rsqrtf(var1 + 1e-5f);

    float4 g = ((const float4*)ln_g)[tid];
    float4 b = ((const float4*)ln_b)[tid];

    float n0x = (xv.x - mu0) * rs0 * g.x + b.x;
    float n0y = (xv.y - mu0) * rs0 * g.y + b.y;
    float n0z = (xv.z - mu0) * rs0 * g.z + b.z;
    float n0w = (xv.w - mu0) * rs0 * g.w + b.w;

    float n1x = 0.f, n1y = 0.f, n1z = 0.f, n1w = 0.f;
    if (t > 0) {
        n1x = (pv.x - mu1) * rs1 * g.x + b.x;
        n1y = (pv.y - mu1) * rs1 * g.y + b.y;
        n1z = (pv.z - mu1) * rs1 * g.z + b.z;
        n1w = (pv.w - mu1) * rs1 * g.w + b.w;
    }

    ushort4 o;
    o.x = f2bf(0.5f * (n0x + n1x));
    o.y = f2bf(0.5f * (n0y + n1y));
    o.z = f2bf(0.5f * (n0z + n1z));
    o.w = f2bf(0.5f * (n0w + n1w));
    ((ushort4*)(Abf + (size_t)bt * C))[tid] = o;
}

// ---------------------------------------------------------------------------
// WMMA bf16 GEMM:  out[m][n] = sum_k A[m][k] * W[n][k]   (i.e. A @ W^T)
// Block = 256 threads = 8 waves laid out 4(M) x 2(N); block tile 128 x 128.
// Wave tile 32(M) x 64(N): 2 A-fragments, 4 B-fragments, 8 WMMAs per k-step.
// Copy-free ping-pong pipelining: two fixed fragment sets f0/f1; K stepped by
// 64 with the tail peeled so the steady-state body has no branches and each
// WMMA group's operands were loaded one stage earlier (loads overlap math).
// mode 0: Ntot = 6*1024, epilogue per column block -> proj[which][M][1024]
//         which: 0=r 1=k 2=v (id), 3=w -> exp(-exp(x)), 4=a 5=g -> sigmoid
// mode 1: Ntot = 1024, identity, row-major out (the Wo projection)
// ---------------------------------------------------------------------------
__global__ __launch_bounds__(256)
void wmma_gemm_kernel(const unsigned short* __restrict__ A,
                      const unsigned short* __restrict__ W,
                      float* __restrict__ out,
                      int M, int Ntot, int K, int mode) {
    const int lane = threadIdx.x & 31;
    const int wid  = threadIdx.x >> 5;
    const int wm   = wid & 3;                   // 0..3 -> M
    const int wn   = wid >> 2;                  // 0..1 -> N
    const int tileM = blockIdx.x * 128 + wm * 32;
    const int tileN = blockIdx.y * 128 + wn * 64;

    const int hf  = lane >> 4;                  // lane half (0/1)
    const int l15 = lane & 15;
    const int akc = hf * 16;                    // A: contiguous K [akc..+15]
    const int bkc = hf * 16;                    // B: contiguous K [bkc..+15]

    v8f z = {};
    v8f acc[8];
    #pragma unroll
    for (int q = 0; q < 8; ++q) acc[q] = z;

    const unsigned short* arow0 = A + (size_t)(tileM + l15) * K;
    const unsigned short* arow1 = arow0 + (size_t)16 * K;
    const unsigned short* brow0 = W + (size_t)(tileN + l15) * K;
    const unsigned short* brow1 = brow0 + (size_t)16 * K;
    const unsigned short* brow2 = brow0 + (size_t)32 * K;
    const unsigned short* brow3 = brow0 + (size_t)48 * K;

    BF16Frag f0[6], f1[6];

    auto load_frags = [&](BF16Frag* f, int kt) {
        f[0].h[0] = *(const v8bf*)(arow0 + kt + akc);
        f[0].h[1] = *(const v8bf*)(arow0 + kt + akc + 8);
        f[1].h[0] = *(const v8bf*)(arow1 + kt + akc);
        f[1].h[1] = *(const v8bf*)(arow1 + kt + akc + 8);
        f[2].h[0] = *(const v8bf*)(brow0 + kt + bkc);
        f[2].h[1] = *(const v8bf*)(brow0 + kt + bkc + 8);
        f[3].h[0] = *(const v8bf*)(brow1 + kt + bkc);
        f[3].h[1] = *(const v8bf*)(brow1 + kt + bkc + 8);
        f[4].h[0] = *(const v8bf*)(brow2 + kt + bkc);
        f[4].h[1] = *(const v8bf*)(brow2 + kt + bkc + 8);
        f[5].h[0] = *(const v8bf*)(brow3 + kt + bkc);
        f[5].h[1] = *(const v8bf*)(brow3 + kt + bkc + 8);
    };

    auto do_wmmas = [&](BF16Frag* f) {
        #pragma unroll
        for (int j = 0; j < 4; ++j) {
            acc[j] = __builtin_amdgcn_wmma_f32_16x16x32_bf16(
                false, f[0].v, false, f[2 + j].v, (short)0, acc[j], false, false);
        }
        #pragma unroll
        for (int j = 0; j < 4; ++j) {
            acc[4 + j] = __builtin_amdgcn_wmma_f32_16x16x32_bf16(
                false, f[1].v, false, f[2 + j].v, (short)0, acc[4 + j], false, false);
        }
    };

    load_frags(f0, 0);

    int kt = 0;
    for (; kt + 64 < K; kt += 64) {             // steady state: branch-free body
        load_frags(f1, kt + 32);
        __builtin_prefetch(arow0 + kt + 512, 0, 1);
        __builtin_prefetch(brow0 + kt + 512, 0, 1);
        do_wmmas(f0);
        load_frags(f0, kt + 64);
        do_wmmas(f1);
    }
    // tail: kt == K - 64
    load_frags(f1, kt + 32);
    do_wmmas(f0);
    do_wmmas(f1);

    // Epilogue. C/D layout: VGPR e, lanes 0-15 -> M = e, N = lane;
    //                              lanes 16-31 -> M = e+8, N = lane-16.
    const int mrow = hf * 8;
    #pragma unroll
    for (int q = 0; q < 8; ++q) {
        const int i = q >> 2, j = q & 3;        // i: M subtile, j: N subtile
        const int nn = tileN + j * 16 + l15;
        #pragma unroll
        for (int e = 0; e < 8; ++e) {
            const int mm = tileM + i * 16 + mrow + e;
            float val = acc[q][e];
            if (mode == 0) {
                const int which = nn >> 10;
                const int c     = nn & 1023;
                if (which == 3)      val = __expf(-__expf(val));         // decay
                else if (which >= 4) val = 1.0f / (1.0f + __expf(-val)); // sigmoid
                out[((size_t)which * M + mm) * 1024 + c] = val;
            } else {
                out[(size_t)mm * 1024 + nn] = val;
            }
        }
    }
}

// ---------------------------------------------------------------------------
// Sequential RWKV7 scan.  One 256-thread block per (b,h).
// Thread (i, q): row i (0..63), quarter q (0..3) owning j in [q*16, q*16+16).
//   sa[i] = <s[i,:], a>   (4-lane __shfl_xor reduction, lanes i*4+q contiguous)
//   s[i][j] = s[i][j]*dec[i] + sa[i] + k[i]*v[j];  y[i] = <s[i,:], r>
// GroupNorm statistics (per (b,h) over T x N) accumulated on q==0 lanes.
// ---------------------------------------------------------------------------
__global__ __launch_bounds__(256)
void rwkv_scan_kernel(const float* __restrict__ proj,      // [6][B*T][C]
                      const float* __restrict__ state_in,  // [B,H,N,N]
                      float* __restrict__ y,               // [B*T*C]
                      float* __restrict__ stats,           // [B*H][2]
                      float* __restrict__ state_out,       // [B,H,N,N]
                      int B, int T, int H) {
    const int N = 64;
    const int C = H * N;
    const int bh = blockIdx.x;
    const int b = bh / H, h = bh % H;
    const int tid = threadIdx.x;
    const int i = tid >> 2;                    // state row
    const int q = tid & 3;                     // quarter of the row
    const int j0 = q * 16;
    const size_t MC = (size_t)B * T * C;

    const float* Pr = proj;
    const float* Pk = proj + MC;
    const float* Pv = proj + 2 * MC;
    const float* Pw = proj + 3 * MC;           // decay = exp(-exp(w_raw))
    const float* Pa = proj + 4 * MC;

    float s[16];
    {
        const float* st = state_in + ((size_t)bh * N + i) * N + j0;
        #pragma unroll
        for (int j = 0; j < 16; ++j) s[j] = st[j];
    }

    __shared__ float a_s[64], v_s[64], r_s[64], k_s[64], w_s[64];
    float sumy = 0.f, sumy2 = 0.f;

    for (int t = 0; t < T; ++t) {
        const size_t base = ((size_t)b * T + t) * C + (size_t)h * N;
        if (tid < 64) {
            r_s[tid] = Pr[base + tid];
            k_s[tid] = Pk[base + tid];
            v_s[tid] = Pv[base + tid];
            w_s[tid] = Pw[base + tid];
            a_s[tid] = Pa[base + tid];
        }
        __syncthreads();

        float sa = 0.f;
        #pragma unroll
        for (int j = 0; j < 16; ++j) sa += s[j] * a_s[j0 + j];
        sa += __shfl_xor(sa, 1, 32);           // reduce across the 4 lanes
        sa += __shfl_xor(sa, 2, 32);           // owning this row

        const float dec = w_s[i];
        const float ki  = k_s[i];
        float yp = 0.f;
        #pragma unroll
        for (int j = 0; j < 16; ++j) {
            s[j] = s[j] * dec + sa + ki * v_s[j0 + j];
            yp  += s[j] * r_s[j0 + j];
        }
        yp += __shfl_xor(yp, 1, 32);
        yp += __shfl_xor(yp, 2, 32);

        if (q == 0) {
            y[base + i] = yp;
            sumy  += yp;
            sumy2 += yp * yp;
        }
        __syncthreads();
    }

    // reduce GN statistics over the 64 rows (q==0 lanes hold them)
    __shared__ float rs[64], rq[64];
    if (q == 0) { rs[i] = sumy; rq[i] = sumy2; }
    __syncthreads();
    for (int off = 32; off > 0; off >>= 1) {
        if (tid < off) { rs[tid] += rs[tid + off]; rq[tid] += rq[tid + off]; }
        __syncthreads();
    }
    if (tid == 0) { stats[bh * 2] = rs[0]; stats[bh * 2 + 1] = rq[0]; }

    float* so = state_out + ((size_t)bh * N + i) * N + j0;
    #pragma unroll
    for (int j = 0; j < 16; ++j) so[j] = s[j];
}

// ---------------------------------------------------------------------------
// GroupNorm (per (b,h) over T x N) + gate, emit bf16 for the Wo GEMM.
// ---------------------------------------------------------------------------
__global__ __launch_bounds__(256)
void gn_gate_kernel(const float* __restrict__ y,
                    const float* __restrict__ gproj,   // sigmoid(x_in @ Wg^T)
                    const float* __restrict__ stats,
                    const float* __restrict__ gn_g,
                    const float* __restrict__ gn_b,
                    unsigned short* __restrict__ A2,
                    int B, int T, int H) {
    const int N = 64, C = H * N;
    const int idx = blockIdx.x * 256 + threadIdx.x;   // < B*T*C = 8388608
    const int c  = idx % C;
    const int bt = idx / C;
    const int b  = bt / T;
    const int h  = c >> 6;
    const int bh = b * H + h;

    const float cnt  = (float)(T * N);
    const float mean = stats[bh * 2] / cnt;
    const float var  = stats[bh * 2 + 1] / cnt - mean * mean;
    const float rstd = rsqrtf(var + 1e-5f);

    const float yn = (y[idx] - mean) * rstd * gn_g[c] + gn_b[c];
    A2[idx] = f2bf(yn * gproj[idx]);
}

// ---------------------------------------------------------------------------
extern "C" void kernel_launch(void* const* d_in, const int* in_sizes, int n_in,
                              void* d_out, int out_size, void* d_ws, size_t ws_size,
                              hipStream_t stream) {
    const int B = 8, T = 1024, C = 1024, H = 16;
    const int M = B * T;                 // 8192
    const int CC = C * C;                // 1048576

    const float* x     = (const float*)d_in[0];
    const float* state = (const float*)d_in[1];
    const float* ln_g  = (const float*)d_in[2];
    const float* ln_b  = (const float*)d_in[3];
    const float* Wr    = (const float*)d_in[4];
    const float* Wk    = (const float*)d_in[5];
    const float* Wv    = (const float*)d_in[6];
    const float* Ww    = (const float*)d_in[7];
    const float* Wg    = (const float*)d_in[8];
    const float* Wa    = (const float*)d_in[9];
    const float* Wo    = (const float*)d_in[10];
    const float* gn_g  = (const float*)d_in[11];
    const float* gn_b  = (const float*)d_in[12];

    float* out       = (float*)d_out;                 // [M*C]
    float* state_out = out + (size_t)M * C;           // [B*H*N*N]

    // workspace layout (all sizes 256B aligned)
    char* p = (char*)d_ws;
    unsigned short* Abf = (unsigned short*)p;  p += (size_t)M * C * 2;       // 16 MB
    unsigned short* W6  = (unsigned short*)p;  p += (size_t)6 * CC * 2;      // 12 MB
    unsigned short* Wob = (unsigned short*)p;  p += (size_t)CC * 2;          //  2 MB
    float* proj  = (float*)p;                  p += (size_t)6 * M * C * 4;   // 192 MB
    float* ybuf  = (float*)p;                  p += (size_t)M * C * 4;       // 32 MB
    float* stats = (float*)p;                  p += 1024;
    unsigned short* A2  = (unsigned short*)p;  p += (size_t)M * C * 2;       // 16 MB

    // 1) weight conversion; pack order: r,k,v,w,a,g  (note Wg/Wa input order!)
    const int cblk = CC / (256 * 4);   // 1024 blocks, 4 elems/thread
    f32_to_bf16_kernel<<<cblk, 256, 0, stream>>>(Wr, W6 + (size_t)0 * CC, CC / 4);
    f32_to_bf16_kernel<<<cblk, 256, 0, stream>>>(Wk, W6 + (size_t)1 * CC, CC / 4);
    f32_to_bf16_kernel<<<cblk, 256, 0, stream>>>(Wv, W6 + (size_t)2 * CC, CC / 4);
    f32_to_bf16_kernel<<<cblk, 256, 0, stream>>>(Ww, W6 + (size_t)3 * CC, CC / 4);
    f32_to_bf16_kernel<<<cblk, 256, 0, stream>>>(Wa, W6 + (size_t)4 * CC, CC / 4);
    f32_to_bf16_kernel<<<cblk, 256, 0, stream>>>(Wg, W6 + (size_t)5 * CC, CC / 4);
    f32_to_bf16_kernel<<<cblk, 256, 0, stream>>>(Wo, Wob, CC / 4);

    // 2) LayerNorm + time-shift -> bf16 activations
    ln_shift_kernel<<<M, 256, 0, stream>>>(x, ln_g, ln_b, Abf, T, C);

    // 3) fused 6-way projection GEMM (M x 6144 = A @ W6^T), WMMA bf16
    {
        dim3 grid(M / 128, (6 * C) / 128);
        wmma_gemm_kernel<<<grid, 256, 0, stream>>>(Abf, W6, proj, M, 6 * C, C, 0);
    }

    // 4) sequential scan (writes y, final state, GN stats)
    rwkv_scan_kernel<<<B * H, 256, 0, stream>>>(proj, state, ybuf, stats,
                                                state_out, B, T, H);

    // 5) GroupNorm + gate -> bf16
    gn_gate_kernel<<<(M * C) / 256, 256, 0, stream>>>(
        ybuf, proj + (size_t)5 * M * C, stats, gn_g, gn_b, A2, B, T, H);

    // 6) output projection GEMM (M x 1024 = A2 @ Wo^T), WMMA bf16
    {
        dim3 grid(M / 128, C / 128);
        wmma_gemm_kernel<<<grid, 256, 0, stream>>>(A2, Wob, out, M, C, C, 1);
    }
}
